// GNNLSTM_15204184228137
// MI455X (gfx1250) — compile-verified, matrix operand order checked by reading, not run on hardware
//
#include <hip/hip_runtime.h>

typedef __attribute__((ext_vector_type(16))) _Float16 v16h;
typedef __attribute__((ext_vector_type(8)))  _Float16 v8h;
typedef __attribute__((ext_vector_type(8)))  float    v8f;

union V16 { v16h v; v8h h[2]; };

#define B_  128
#define S_  2048
#define H_  64
#define CH_ 32    // xbar prefetch chunk (steps); must be even

__device__ __forceinline__ float fast_tanh(float x) {
#if __has_builtin(__builtin_amdgcn_tanhf)
  return __builtin_amdgcn_tanhf(x);   // gfx1250 v_tanh_f32
#else
  return tanhf(x);
#endif
}
__device__ __forceinline__ float fast_sigmoid(float x) {
  return 0.5f * fast_tanh(0.5f * x) + 0.5f;
}

// xbarP[b/16][s][b%16] = mean over 3 nodes of x[b][s][:]  (per-WG contiguous slab)
__global__ void gl_prep_x(const float* __restrict__ x, float* __restrict__ xbarP) {
  int idx = blockIdx.x * blockDim.x + threadIdx.x;  // B_*S_ threads exactly
  int b = idx & (B_ - 1);
  int s = idx >> 7;
  const float* p = x + ((size_t)b * S_ + s) * 3;
  xbarP[(size_t)(b >> 4) * (S_ * 16) + s * 16 + (b & 15)] =
      (p[0] + p[1] + p[2]) * (1.0f / 3.0f);
}

// Gate-permuted precompute: n' = j*4 + gate  (gate in {i,f,g,o} = n/64, j = n%64)
// u'[n'] = w_ih[n,:] @ gnn_w ; v'[n'] = w_ih[n,:] @ gnn_b + b_ih[n] + b_hh[n]
// whh16[n',k] = (f16) w_hh[n,k]
__global__ void gl_prep_w(const float* __restrict__ w_ih, const float* __restrict__ gnn_w,
                          const float* __restrict__ gnn_b, const float* __restrict__ b_ih,
                          const float* __restrict__ b_hh, const float* __restrict__ w_hh,
                          float* __restrict__ u, float* __restrict__ v,
                          _Float16* __restrict__ whh16) {
  int n = threadIdx.x;          // source (gate-major) index, 0..255
  int g = n >> 6, j = n & 63;
  int np = j * 4 + g;           // permuted index
  float su = 0.f, sv = 0.f;
  for (int k = 0; k < H_; ++k) {
    float w = w_ih[n * H_ + k];
    su += w * gnn_w[k];
    sv += w * gnn_b[k];
  }
  u[np] = su;
  v[np] = sv + b_ih[n] + b_hh[n];
  for (int k = 0; k < H_; ++k)
    whh16[np * H_ + k] = (_Float16)w_hh[n * H_ + k];
}

// One WG = 16 batch rows, 8 waves. Wave w owns permuted gate cols [32w,32w+32)
// == hidden units j in [8w, 8w+8) with all 4 gates adjacent.
__global__ __launch_bounds__(256, 1)
void gl_lstm(const float* __restrict__ xbarP, const _Float16* __restrict__ whh,
             const float* __restrict__ u, const float* __restrict__ v,
             const float* __restrict__ fc_w, const float* __restrict__ fc_b,
             float* __restrict__ out) {
  __shared__ __align__(16) float    wgate[8 * 16 * 32];   // 16 KB, per-wave private tiles
  __shared__ __align__(16) _Float16 hbuf[2][16 * H_];     // 4 KB, double-buffered h (f16)
  __shared__ __align__(16) float    xb_buf[2][CH_ * 16];  // 4 KB, xbar chunk double buffer

  const int tid  = threadIdx.x;
  const int wave = tid >> 5;     // wave32
  const int lane = tid & 31;
  const int b0   = blockIdx.x * 16;
  const float* slab = xbarP + (size_t)blockIdx.x * (S_ * 16);

  for (int i = tid; i < 16 * H_; i += 256) hbuf[0][i] = (_Float16)0.0f;

  // Resident B fragments of permuted w_hh: 2 N-tiles x 2 K-halves per wave.
  const int lrow = lane & 15;          // M (A) / N (B) within tile
  const int kb   = (lane >> 4) * 8;    // K sub-chunk base per half-wave
  V16 bfrag[2][2];
#pragma unroll
  for (int ti = 0; ti < 2; ++ti) {
    int n = (wave * 2 + ti) * 16 + lrow;
#pragma unroll
    for (int kh = 0; kh < 2; ++kh) {
      int off = kh * 32 + kb;
      bfrag[ti][kh].h[0] = *(const v8h*)(whh + n * H_ + off);
      bfrag[ti][kh].h[1] = *(const v8h*)(whh + n * H_ + off + 16);
    }
  }

  // Element-wise ownership: lane -> hidden unit j, 4 batch rows m = mrow*4+mm.
  const int j_loc = lane & 7;
  const int mrow  = lane >> 3;          // 0..3
  const int ew_j  = wave * 8 + j_loc;
  float4 uq = *(const float4*)(u + ew_j * 4);   // u' quad (i,f,g,o), resident
  float4 vq = *(const float4*)(v + ew_j * 4);
  float c[4] = {0.f, 0.f, 0.f, 0.f};

  const int wb    = wave * 512;         // wave's private wgate region (floats)
  const int rbase = (lane >> 4) * 8;    // C rows: M = r (+8 for upper half-wave)

  // One LSTM step: h(src) -> gates -> h(dst); one barrier. Branch-free body.
  auto lstm_step = [&](const _Float16* __restrict__ hsrc, _Float16* __restrict__ hdst,
                       const float* __restrict__ xbrow) {
    V16 afrag[2];
#pragma unroll
    for (int kh = 0; kh < 2; ++kh) {
      int off = kh * 32 + kb;
      afrag[kh].h[0] = *(const v8h*)(hsrc + lrow * H_ + off);
      afrag[kh].h[1] = *(const v8h*)(hsrc + lrow * H_ + off + 16);
    }
#pragma unroll
    for (int ti = 0; ti < 2; ++ti) {
      v8f acc = {};
      acc = __builtin_amdgcn_wmma_f32_16x16x32_f16(false, afrag[0].v, false, bfrag[ti][0].v,
                                                   (short)0, acc, false, false);
      acc = __builtin_amdgcn_wmma_f32_16x16x32_f16(false, afrag[1].v, false, bfrag[ti][1].v,
                                                   (short)0, acc, false, false);
      int nl = ti * 16 + lrow;
#pragma unroll
      for (int r = 0; r < 8; ++r)
        wgate[wb + (rbase + r) * 32 + nl] = acc[r];
    }
    // Same-wave transpose read-back (DS in-order within wave; no barrier needed).
    float xbv[4];
    *(float4*)xbv = *(const float4*)(xbrow + mrow * 4);
#pragma unroll
    for (int mm = 0; mm < 4; ++mm) {
      int   m = mrow * 4 + mm;
      float gv[4];
      *(float4*)gv = *(const float4*)&wgate[wb + m * 32 + j_loc * 4];
      float xb = xbv[mm];
      float gi = gv[0] + xb * uq.x + vq.x;
      float gf = gv[1] + xb * uq.y + vq.y;
      float gg = gv[2] + xb * uq.z + vq.z;
      float go = gv[3] + xb * uq.w + vq.w;
      float ci = fast_sigmoid(gi);
      float cf = fast_sigmoid(gf);
      float co = fast_sigmoid(go);
      float cg = fast_tanh(gg);
      c[mm] = cf * c[mm] + ci * cg;
      float hn = co * fast_tanh(c[mm]);
      hdst[m * H_ + ew_j] = (_Float16)hn;
    }
    __syncthreads();
  };

  // 2-deep xbar software pipeline: chunk tc is in LDS, tc+1 in registers,
  // tc+2 in flight from global. All staging at chunk granularity.
  const int NCH = S_ / CH_;
  *(float2*)&xb_buf[0][tid * 2] = *(const float2*)(slab + tid * 2);          // chunk 0
  float2 pf = *(const float2*)(slab + (CH_ * 16) + tid * 2);                 // chunk 1
  __syncthreads();

  for (int tc = 0; tc < NCH; ++tc) {
    if (tc + 1 < NCH) *(float2*)&xb_buf[(tc + 1) & 1][tid * 2] = pf;  // stage tc+1
    if (tc + 2 < NCH) pf = *(const float2*)(slab + (tc + 2) * (CH_ * 16) + tid * 2);
    const float* xb = xb_buf[tc & 1];
#pragma unroll 1
    for (int ts = 0; ts < CH_; ts += 2) {   // even chunk => compile-time h buffers
      lstm_step(hbuf[0], hbuf[1], xb + ts * 16);
      lstm_step(hbuf[1], hbuf[0], xb + (ts + 1) * 16);
    }
  }

  // out[b, :] = h_last @ fc_w.T + fc_b   (OUT_DIM=3); h_last is in hbuf[0].
  if (tid < 48) {
    int m = tid / 3, o = tid % 3;
    float s = fc_b[o];
    for (int k = 0; k < H_; ++k) s += (float)hbuf[0][m * H_ + k] * fc_w[o * H_ + k];
    out[(b0 + m) * 3 + o] = s;
  }
}

extern "C" void kernel_launch(void* const* d_in, const int* in_sizes, int n_in,
                              void* d_out, int out_size, void* d_ws, size_t ws_size,
                              hipStream_t stream) {
  const float* x     = (const float*)d_in[0];
  const float* gnn_w = (const float*)d_in[1];
  const float* gnn_b = (const float*)d_in[2];
  const float* w_ih  = (const float*)d_in[3];
  const float* w_hh  = (const float*)d_in[4];
  const float* b_ih  = (const float*)d_in[5];
  const float* b_hh  = (const float*)d_in[6];
  const float* fc_w  = (const float*)d_in[7];
  const float* fc_b  = (const float*)d_in[8];
  float* out = (float*)d_out;

  char* ws = (char*)d_ws;
  float*    xbarP = (float*)ws;                               // 1 MB  [B/16][S][16]
  _Float16* whh16 = (_Float16*)(ws + (1u << 20));             // 32 KB (gate-permuted)
  float*    u     = (float*)(ws + (1u << 20) + 32768);        // 1 KB  (gate-permuted)
  float*    v     = (float*)(ws + (1u << 20) + 32768 + 1024); // 1 KB  (gate-permuted)

  gl_prep_x<<<(B_ * S_) / 256, 256, 0, stream>>>(x, xbarP);
  gl_prep_w<<<1, 256, 0, stream>>>(w_ih, gnn_w, gnn_b, b_ih, b_hh, w_hh, u, v, whh16);
  gl_lstm<<<8, 256, 0, stream>>>(xbarP, whh16, u, v, fc_w, fc_b, out);
}
